// EquivariantGraphNetwork_35974646072148
// MI455X (gfx1250) — compile-verified, matrix operand order checked by reading, not run on hardware
//
#include <hip/hip_runtime.h>
#include <hip/hip_bf16.h>

// ---------------------------------------------------------------------------
// EGNN forward, fused for gfx1250 (wave32, WMMA f16 16x16x32, f32 accum).
// Output depends only on the h-path; the coord update in the reference is
// dead code and is eliminated. Each wave owns a 32-row tile (2 WMMA M-tiles)
// so every B fragment feeds 2 WMMAs -> half the L2 weight traffic and a
// deeper load->wmma pipeline.
// ---------------------------------------------------------------------------

typedef __attribute__((ext_vector_type(16))) _Float16 v16h;
typedef __attribute__((ext_vector_type(8)))  float    v8f;

#define H 128

__device__ __forceinline__ float silu_f(float x) {
    return x / (1.0f + __expf(-x));
}

__device__ __forceinline__ void atomAddF(float* p, float v) {
    unsafeAtomicAdd(p, v);
}

// Load a v16h (32B) from two 16B chunks.
__device__ __forceinline__ v16h ld_v16h(const _Float16* p0, const _Float16* p1) {
    union { uint4 q[2]; v16h v; } u;
    u.q[0] = *(const uint4*)p0;
    u.q[1] = *(const uint4*)p1;
    return u.v;
}

// A-fragment (16x32 f16) for k-chunk kc from row-major [16][ldk] f16 tile.
// ISA 7.12.2: lanes 0-15 (M=lane) hold K {0..7,16..23}; lanes 16-31
// (M=lane-16) hold K {8..15,24..31}, 2 halfs per VGPR.
__device__ __forceinline__ v16h load_afrag(const _Float16* sA, int ldk, int lane, int kc) {
    int m  = lane & 15;
    int kb = kc * 32 + ((lane & 16) ? 8 : 0);
    const _Float16* base = sA + m * ldk + kb;
    return ld_v16h(base, base + 16);
}

// B-fragment from pre-packed weights: fragment f = 512 contiguous halfs;
// lane L reads halfs [L*16 .. L*16+15].
__device__ __forceinline__ v16h load_bfrag(const _Float16* pk, int frag, int lane) {
    const _Float16* base = pk + (size_t)frag * 512 + lane * 16;
    return ld_v16h(base, base + 8);
}

// 32xK @ Kx128 GEMM: two M-tiles share each B fragment. acc initialized with
// bias (C layout: element (M,N) at VGPR r: M = r + (lane<16?0:8),
// N = nc*16 + (lane&15)).
template <int NKC>
__device__ __forceinline__ void gemm32(const _Float16* sA, int ldk,
                                       const _Float16* pkB, const float* bias,
                                       v8f (&acc)[2][8], int lane) {
    int ln = lane & 15;
#pragma unroll
    for (int nc = 0; nc < 8; ++nc) {
        float b = bias[nc * 16 + ln];
#pragma unroll
        for (int r = 0; r < 8; ++r) { acc[0][nc][r] = b; acc[1][nc][r] = b; }
    }
#pragma unroll
    for (int kc = 0; kc < NKC; ++kc) {
        v16h a0 = load_afrag(sA, ldk, lane, kc);
        v16h a1 = load_afrag(sA + 16 * ldk, ldk, lane, kc);
#pragma unroll
        for (int nc = 0; nc < 8; ++nc) {
            v16h b = load_bfrag(pkB, kc * 8 + nc, lane);
            acc[0][nc] = __builtin_amdgcn_wmma_f32_16x16x32_f16(
                false, a0, false, b, (short)0, acc[0][nc], false, false);
            acc[1][nc] = __builtin_amdgcn_wmma_f32_16x16x32_f16(
                false, a1, false, b, (short)0, acc[1][nc], false, false);
        }
    }
}

// C fragment (one 16-row tile) -> SiLU -> f16 row-major [16][ldo] LDS tile.
__device__ __forceinline__ void silu_store16(_Float16* sOut, int ldo, int lane, v8f (&acc)[8]) {
    int ln = lane & 15, mb = (lane & 16) ? 8 : 0;
#pragma unroll
    for (int nc = 0; nc < 8; ++nc)
#pragma unroll
        for (int r = 0; r < 8; ++r)
            sOut[(mb + r) * ldo + nc * 16 + ln] = (_Float16)silu_f(acc[nc][r]);
}

__device__ __forceinline__ void silu_store32(_Float16* sOut, int ldo, int lane, v8f (&acc)[2][8]) {
    silu_store16(sOut, ldo, lane, acc[0]);
    silu_store16(sOut + 16 * ldo, ldo, lane, acc[1]);
}

// ---------------------------------------------------------------------------
// Weight packing: W [K x 128] f32 (row-major, jax x@W layout) -> f16 WMMA
// B-fragments, [kc][nc] fragment order, K zero-padded to nKc*32.
// ---------------------------------------------------------------------------
__global__ void pack_b_kernel(const float* __restrict__ W, _Float16* __restrict__ dst,
                              int Kreal, int nKc) {
    int t = blockIdx.x * blockDim.x + threadIdx.x;
    int total = nKc * 8 * 512;
    if (t >= total) return;
    int frag = t >> 9;
    int s    = t & 511;
    int kc = frag >> 3, nc = frag & 7;
    int lane = s >> 4, h = s & 15;
    int v = h >> 1, part = h & 1;
    int n   = nc * 16 + (lane & 15);
    int kin = ((v < 4) ? 0 : 16) + ((lane >= 16) ? 8 : 0) + ((v & 3) << 1) + part;
    int k   = kc * 32 + kin;
    float val = (k < Kreal) ? W[(size_t)k * 128 + n] : 0.0f;
    dst[t] = (_Float16)val;
}

__global__ void zero_kernel(float* __restrict__ m_agg, float* __restrict__ pooled,
                            int n_magg, int n_pooled) {
    int i = blockIdx.x * blockDim.x + threadIdx.x;
    if (i < n_magg)   m_agg[i]  = 0.0f;
    if (i < n_pooled) pooled[i] = 0.0f;
}

// ---------------------------------------------------------------------------
// h = silu(nodes @ emb_w + b) ; store f32 (for residual) and f16 (for gather)
// ---------------------------------------------------------------------------
__global__ __launch_bounds__(128) void embed_kernel(
    const float* __restrict__ nodes, const _Float16* __restrict__ pk_emb,
    const float* __restrict__ b_emb, float* __restrict__ hf32,
    _Float16* __restrict__ hf16, int N) {
    __shared__ _Float16 smem[4][32 * 64];
    int lane = threadIdx.x & 31, wave = threadIdx.x >> 5;
    int tile = blockIdx.x * 4 + wave;
    if (tile * 32 >= N) return;
    _Float16* sm = smem[wave];
    int nd = tile * 32 + lane;
    {
        const float4* src = (const float4*)(nodes + (size_t)nd * 64);
#pragma unroll
        for (int j = 0; j < 16; ++j) {
            float4 x = src[j];
            sm[lane * 64 + 4 * j + 0] = (_Float16)x.x;
            sm[lane * 64 + 4 * j + 1] = (_Float16)x.y;
            sm[lane * 64 + 4 * j + 2] = (_Float16)x.z;
            sm[lane * 64 + 4 * j + 3] = (_Float16)x.w;
        }
    }
    v8f acc[2][8];
    gemm32<2>(sm, 64, pk_emb, b_emb, acc, lane);

    int ln = lane & 15, mb = (lane & 16) ? 8 : 0;
#pragma unroll
    for (int t = 0; t < 2; ++t)
#pragma unroll
        for (int nc = 0; nc < 8; ++nc)
#pragma unroll
            for (int r = 0; r < 8; ++r) {
                float y = silu_f(acc[t][nc][r]);
                size_t idx = (size_t)(tile * 32 + t * 16 + mb + r) * H + nc * 16 + ln;
                hf32[idx] = y;
                hf16[idx] = (_Float16)y;
            }
}

// ---------------------------------------------------------------------------
// Fused edge chain: m_in -> edge0 -> silu -> edge1 -> silu = m
//                   m -> att0 -> silu -> att1 -> sigmoid = att
//                   atomic scatter of m*att into m_agg[row]
// One wave per 32-edge tile; one edge per lane for staging/scatter; all
// staging in wave-private LDS (same-wave DS ops are in-order -> no barriers).
// ---------------------------------------------------------------------------
__global__ __launch_bounds__(128) void edge_kernel(
    const _Float16* __restrict__ hf16, const float* __restrict__ coord,
    const int* __restrict__ edges, const float* __restrict__ edge_attr,
    const _Float16* __restrict__ pk_edge0, const float* __restrict__ b_edge0,
    const _Float16* __restrict__ pk_edge1, const float* __restrict__ b_edge1,
    const _Float16* __restrict__ pk_att0,  const float* __restrict__ b_att0,
    const float* __restrict__ w_att1, const float* __restrict__ b_att1,
    float* __restrict__ m_agg, int E) {
    __shared__ _Float16 smem[4][32 * 320];
    int lane = threadIdx.x & 31, wave = threadIdx.x >> 5;
    int tile = blockIdx.x * 4 + wave;
    if (tile * 32 >= E) return;
    _Float16* sm = smem[wave];

    int eg = tile * 32 + lane;
    int r0 = edges[eg];
    int c0 = edges[E + eg];

    // Stage m_in [32][320]: [h[row] | h[col] | radial | edge_attr | pad]
    {
        const uint4* hr = (const uint4*)(hf16 + (size_t)r0 * H);
        const uint4* hc = (const uint4*)(hf16 + (size_t)c0 * H);
        uint4* dA = (uint4*)(sm + lane * 320);
#pragma unroll
        for (int j = 0; j < 16; ++j) dA[j] = hr[j];
#pragma unroll
        for (int j = 0; j < 16; ++j) dA[16 + j] = hc[j];
        float dx = coord[r0 * 3 + 0] - coord[c0 * 3 + 0];
        float dy = coord[r0 * 3 + 1] - coord[c0 * 3 + 1];
        float dz = coord[r0 * 3 + 2] - coord[c0 * 3 + 2];
        float radial = dx * dx + dy * dy + dz * dz;
        sm[lane * 320 + 256] = (_Float16)radial;
        const float4* ea = (const float4*)(edge_attr + (size_t)eg * 32);
#pragma unroll
        for (int j = 0; j < 8; ++j) {
            float4 x = ea[j];
            sm[lane * 320 + 257 + 4 * j + 0] = (_Float16)x.x;
            sm[lane * 320 + 257 + 4 * j + 1] = (_Float16)x.y;
            sm[lane * 320 + 257 + 4 * j + 2] = (_Float16)x.z;
            sm[lane * 320 + 257 + 4 * j + 3] = (_Float16)x.w;
        }
#pragma unroll
        for (int j = 289; j < 320; ++j) sm[lane * 320 + j] = (_Float16)0.0f;
    }

    v8f acc[2][8];
    // edge0: K = 320 (289 padded)
    gemm32<10>(sm, 320, pk_edge0, b_edge0, acc, lane);
    silu_store32(sm, H, lane, acc);                 // act1 at halfs [0, 4096)
    // edge1: K = 128
    gemm32<4>(sm, H, pk_edge1, b_edge1, acc, lane);
    _Float16* sm_m = sm + 32 * H;                   // m at halfs [4096, 8192)
    silu_store32(sm_m, H, lane, acc);
    // att0: K = 128
    gemm32<4>(sm_m, H, pk_att0, b_att0, acc, lane);
    silu_store32(sm, H, lane, acc);                 // t overwrites act1
    // att1: 128 -> 1 matvec, one edge per lane
    float dot = b_att1[0];
#pragma unroll
    for (int c = 0; c < H; ++c)
        dot += (float)sm[lane * H + c] * w_att1[c];
    float att = 1.0f / (1.0f + __expf(-dot));

    // segment_sum(m * att) over rows via f32 L2 atomics
#pragma unroll
    for (int c = 0; c < H; ++c) {
        float v = (float)sm_m[lane * H + c] * att;
        atomAddF(m_agg + (size_t)r0 * H + c, v);
    }
}

// ---------------------------------------------------------------------------
// Fused node chain: n_in -> node0 -> silu -> node1 (+h residual) = h2
//                   h2 -> enc0 -> silu -> enc1 = henc
//                   atomic pool of henc into pooled[batch[node]]
// ---------------------------------------------------------------------------
__global__ __launch_bounds__(128) void node_kernel(
    const _Float16* __restrict__ hf16, const float* __restrict__ hf32,
    const float* __restrict__ m_agg, const float* __restrict__ node_attr,
    const int* __restrict__ batch,
    const _Float16* __restrict__ pk_node0, const float* __restrict__ b_node0,
    const _Float16* __restrict__ pk_node1, const float* __restrict__ b_node1,
    const _Float16* __restrict__ pk_enc0,  const float* __restrict__ b_enc0,
    const _Float16* __restrict__ pk_enc1,  const float* __restrict__ b_enc1,
    float* __restrict__ pooled, int N) {
    __shared__ _Float16 smem[4][32 * 288];
    int lane = threadIdx.x & 31, wave = threadIdx.x >> 5;
    int tile = blockIdx.x * 4 + wave;
    if (tile * 32 >= N) return;
    _Float16* sm = smem[wave];
    int nd = tile * 32 + lane;

    // n_in [32][288] = [h (f16) | agg (f32->f16) | node_attr (f32->f16)]
    {
        const uint4* hr = (const uint4*)(hf16 + (size_t)nd * H);
        uint4* dA = (uint4*)(sm + lane * 288);
#pragma unroll
        for (int j = 0; j < 16; ++j) dA[j] = hr[j];
        const float4* ag = (const float4*)(m_agg + (size_t)nd * H);
#pragma unroll
        for (int j = 0; j < 32; ++j) {
            float4 x = ag[j];
            sm[lane * 288 + 128 + 4 * j + 0] = (_Float16)x.x;
            sm[lane * 288 + 128 + 4 * j + 1] = (_Float16)x.y;
            sm[lane * 288 + 128 + 4 * j + 2] = (_Float16)x.z;
            sm[lane * 288 + 128 + 4 * j + 3] = (_Float16)x.w;
        }
        const float4* na = (const float4*)(node_attr + (size_t)nd * 32);
#pragma unroll
        for (int j = 0; j < 8; ++j) {
            float4 x = na[j];
            sm[lane * 288 + 256 + 4 * j + 0] = (_Float16)x.x;
            sm[lane * 288 + 256 + 4 * j + 1] = (_Float16)x.y;
            sm[lane * 288 + 256 + 4 * j + 2] = (_Float16)x.z;
            sm[lane * 288 + 256 + 4 * j + 3] = (_Float16)x.w;
        }
    }

    v8f acc[2][8];
    gemm32<9>(sm, 288, pk_node0, b_node0, acc, lane);
    silu_store32(sm, H, lane, acc);                 // act1 at halfs [0, 4096)
    gemm32<4>(sm, H, pk_node1, b_node1, acc, lane);

    // residual h2 = h + out; stage f16 at halfs [4096, 8192)
    _Float16* sm2 = sm + 32 * H;
    int ln = lane & 15, mb = (lane & 16) ? 8 : 0;
#pragma unroll
    for (int t = 0; t < 2; ++t)
#pragma unroll
        for (int nc = 0; nc < 8; ++nc)
#pragma unroll
            for (int r = 0; r < 8; ++r) {
                size_t idx = (size_t)(tile * 32 + t * 16 + mb + r) * H + nc * 16 + ln;
                float h2 = hf32[idx] + acc[t][nc][r];
                sm2[(t * 16 + mb + r) * H + nc * 16 + ln] = (_Float16)h2;
            }

    gemm32<4>(sm2, H, pk_enc0, b_enc0, acc, lane);
    silu_store32(sm, H, lane, acc);
    gemm32<4>(sm, H, pk_enc1, b_enc1, acc, lane);

    // global add-pool into pooled[batch[node]]
#pragma unroll
    for (int t = 0; t < 2; ++t)
#pragma unroll
        for (int nc = 0; nc < 8; ++nc)
#pragma unroll
            for (int r = 0; r < 8; ++r) {
                int node = tile * 32 + t * 16 + mb + r;
                int g = batch[node];
                atomAddF(pooled + (size_t)g * H + nc * 16 + ln, acc[t][nc][r]);
            }
}

// ---------------------------------------------------------------------------
// Decoder: pooled -> dec0 -> silu -> dec1 (128->1). G=64 -> 2 waves x 32 rows.
// ---------------------------------------------------------------------------
__global__ __launch_bounds__(64) void dec_kernel(
    const float* __restrict__ pooled, const _Float16* __restrict__ pk_dec0,
    const float* __restrict__ b_dec0, const float* __restrict__ w_dec1,
    const float* __restrict__ b_dec1, float* __restrict__ out, int G) {
    __shared__ _Float16 smem[2][32 * H];
    int lane = threadIdx.x & 31, wave = threadIdx.x >> 5;
    int tile = wave;
    if (tile * 32 >= G) return;
    _Float16* sm = smem[wave];
    int g = tile * 32 + lane;
    {
        const float4* pr = (const float4*)(pooled + (size_t)g * H);
#pragma unroll
        for (int j = 0; j < 32; ++j) {
            float4 x = pr[j];
            sm[lane * H + 4 * j + 0] = (_Float16)x.x;
            sm[lane * H + 4 * j + 1] = (_Float16)x.y;
            sm[lane * H + 4 * j + 2] = (_Float16)x.z;
            sm[lane * H + 4 * j + 3] = (_Float16)x.w;
        }
    }
    v8f acc[2][8];
    gemm32<4>(sm, H, pk_dec0, b_dec0, acc, lane);
    silu_store32(sm, H, lane, acc);

    float dot = b_dec1[0];
#pragma unroll
    for (int c = 0; c < H; ++c)
        dot += (float)sm[lane * H + c] * w_dec1[c];
    out[g] = dot;
}

// ---------------------------------------------------------------------------
// Host-side orchestration
// ---------------------------------------------------------------------------
extern "C" void kernel_launch(void* const* d_in, const int* in_sizes, int n_in,
                              void* d_out, int out_size, void* d_ws, size_t ws_size,
                              hipStream_t stream) {
    const int N = in_sizes[0] / 64;   // 40000
    const int E = in_sizes[2] / 2;    // 640000
    const int G = out_size;           // 64

    const float* nodes     = (const float*)d_in[0];
    const float* coord     = (const float*)d_in[1];
    const int*   edges     = (const int*)  d_in[2];
    const float* edge_attr = (const float*)d_in[3];
    const float* node_attr = (const float*)d_in[4];
    const int*   batch     = (const int*)  d_in[5];
    const float* emb_w   = (const float*)d_in[7],  *emb_b   = (const float*)d_in[8];
    const float* edge0_w = (const float*)d_in[9],  *edge0_b = (const float*)d_in[10];
    const float* edge1_w = (const float*)d_in[11], *edge1_b = (const float*)d_in[12];
    const float* att0_w  = (const float*)d_in[13], *att0_b  = (const float*)d_in[14];
    const float* att1_w  = (const float*)d_in[15], *att1_b  = (const float*)d_in[16];
    // d_in[17..20]: coord0/coord1 — dead code, unused
    const float* node0_w = (const float*)d_in[21], *node0_b = (const float*)d_in[22];
    const float* node1_w = (const float*)d_in[23], *node1_b = (const float*)d_in[24];
    const float* enc0_w  = (const float*)d_in[25], *enc0_b  = (const float*)d_in[26];
    const float* enc1_w  = (const float*)d_in[27], *enc1_b  = (const float*)d_in[28];
    const float* dec0_w  = (const float*)d_in[29], *dec0_b  = (const float*)d_in[30];
    const float* dec1_w  = (const float*)d_in[31], *dec1_b  = (const float*)d_in[32];
    float* out = (float*)d_out;

    // Workspace layout
    char* ws = (char*)d_ws;
    _Float16* pk    = (_Float16*)ws;                     // 184320 halfs packed weights
    float*    hf32  = (float*)(ws + 524288);             // N*128 f32
    _Float16* hf16  = (_Float16*)(ws + 524288 + (size_t)N * H * 4);
    float*    m_agg = (float*)(ws + 524288 + (size_t)N * H * 6);
    float*    pooled= (float*)(ws + 524288 + (size_t)N * H * 10);

    _Float16* pk_emb   = pk;            // nKc=2  ->  8192
    _Float16* pk_edge0 = pk + 8192;     // nKc=10 -> 40960
    _Float16* pk_edge1 = pk + 49152;    // nKc=4
    _Float16* pk_att0  = pk + 65536;    // nKc=4
    _Float16* pk_node0 = pk + 81920;    // nKc=9  -> 36864
    _Float16* pk_node1 = pk + 118784;   // nKc=4
    _Float16* pk_enc0  = pk + 135168;   // nKc=4
    _Float16* pk_enc1  = pk + 151552;   // nKc=4
    _Float16* pk_dec0  = pk + 167936;   // nKc=4

    auto packGrid = [](int nKc) { return dim3((nKc * 4096 + 255) / 256); };
    pack_b_kernel<<<packGrid(2),  256, 0, stream>>>(emb_w,   pk_emb,   64,  2);
    pack_b_kernel<<<packGrid(10), 256, 0, stream>>>(edge0_w, pk_edge0, 289, 10);
    pack_b_kernel<<<packGrid(4),  256, 0, stream>>>(edge1_w, pk_edge1, 128, 4);
    pack_b_kernel<<<packGrid(4),  256, 0, stream>>>(att0_w,  pk_att0,  128, 4);
    pack_b_kernel<<<packGrid(9),  256, 0, stream>>>(node0_w, pk_node0, 288, 9);
    pack_b_kernel<<<packGrid(4),  256, 0, stream>>>(node1_w, pk_node1, 128, 4);
    pack_b_kernel<<<packGrid(4),  256, 0, stream>>>(enc0_w,  pk_enc0,  128, 4);
    pack_b_kernel<<<packGrid(4),  256, 0, stream>>>(enc1_w,  pk_enc1,  128, 4);
    pack_b_kernel<<<packGrid(4),  256, 0, stream>>>(dec0_w,  pk_dec0,  128, 4);

    int n_magg = N * H, n_pooled = G * H;
    zero_kernel<<<(n_magg + 255) / 256, 256, 0, stream>>>(m_agg, pooled, n_magg, n_pooled);

    int nodeTiles = (N + 31) / 32;   // 1250
    int edgeTiles = (E + 31) / 32;   // 20000
    embed_kernel<<<(nodeTiles + 3) / 4, 128, 0, stream>>>(nodes, pk_emb, emb_b,
                                                          hf32, hf16, N);
    edge_kernel<<<(edgeTiles + 3) / 4, 128, 0, stream>>>(
        hf16, coord, edges, edge_attr,
        pk_edge0, edge0_b, pk_edge1, edge1_b, pk_att0, att0_b,
        att1_w, att1_b, m_agg, E);
    node_kernel<<<(nodeTiles + 3) / 4, 128, 0, stream>>>(
        hf16, hf32, m_agg, node_attr, batch,
        pk_node0, node0_b, pk_node1, node1_b,
        pk_enc0, enc0_b, pk_enc1, enc1_b, pooled, N);
    dec_kernel<<<1, 64, 0, stream>>>(pooled, pk_dec0, dec0_b, dec1_w, dec1_b, out, G);
}
